// RNN_image_sentence_parser_61246233641560
// MI455X (gfx1250) — compile-verified, matrix operand order checked by reading, not run on hardware
//
#include <hip/hip_runtime.h>

// Problem constants (match the reference)
#define B_    16
#define S_    512
#define V_    32000
#define D_    512
#define L_    64
#define NSTEP (S_ - 1)
#define ADJW  (S_ / 32)   // 16 x u32 words per adjacency row (512 bits)

typedef __attribute__((ext_vector_type(16))) __bf16 v16bf;
typedef __attribute__((ext_vector_type(8)))  __bf16 v8bf;
typedef __attribute__((ext_vector_type(8)))  float  v8f;

// Load a 16-element bf16 WMMA operand segment pair: [p, p+8) and [p+16, p+24)
// (both 16-byte aligned) -> two b128 loads + shuffle into one v16bf.
__device__ inline v16bf load_frag_bf16(const __bf16* __restrict__ p) {
  const v8bf lo = *(const v8bf*)p;
  const v8bf hi = *(const v8bf*)(p + 16);
  return __builtin_shufflevector(lo, hi, 0, 1, 2, 3, 4, 5, 6, 7,
                                 8, 9, 10, 11, 12, 13, 14, 15);
}

// ---------------------------------------------------------------------------
// Kernel 0: u = W1a @ w2, v = W1b @ w2, c = b1.w2 + b2   (one 512-thread block)
// ---------------------------------------------------------------------------
__global__ __launch_bounds__(512) void uvc_kernel(
    const float* __restrict__ W1, const float* __restrict__ b1,
    const float* __restrict__ w2, const float* __restrict__ b2,
    float* __restrict__ u, float* __restrict__ v, float* __restrict__ cbuf) {
  __shared__ float red[512];
  const int d = threadIdx.x;
  float su = 0.f, sv = 0.f;
  for (int k = 0; k < D_; ++k) {
    const float w = w2[k];
    su = fmaf(W1[(size_t)d * D_ + k], w, su);
    sv = fmaf(W1[(size_t)(D_ + d) * D_ + k], w, sv);
  }
  u[d] = su;
  v[d] = sv;
  red[d] = b1[d] * w2[d];
  __syncthreads();
  for (int s = 256; s > 0; s >>= 1) {
    if (d < s) red[d] += red[d + s];
    __syncthreads();
  }
  if (d == 0) cbuf[0] = red[0] + b2[0];
}

// ---------------------------------------------------------------------------
// Kernel 1: feats = vocab_emb[token_ids]  (coalesced gather into workspace)
// ---------------------------------------------------------------------------
__global__ __launch_bounds__(256) void gather_kernel(
    const int* __restrict__ tok, const float* __restrict__ emb,
    float* __restrict__ fb) {
  const int idx = blockIdx.x * 256 + threadIdx.x;   // B*S*D total
  const int row = idx >> 9;                         // D_ == 512
  const int col = idx & (D_ - 1);
  fb[idx] = emb[(size_t)tok[row] * D_ + col];
}

// ---------------------------------------------------------------------------
// Kernel 1b: posT[n][k] = bf16(pos_W[k][n])  (64 KB, column-major bf16 copy so
// B fragments become contiguous 16-byte loads; stays L2/WGP$-resident)
// ---------------------------------------------------------------------------
__global__ __launch_bounds__(256) void posT_kernel(
    const float* __restrict__ pos_W, __bf16* __restrict__ posT) {
  const int idx = blockIdx.x * 256 + threadIdx.x;   // D*L total
  const int n = idx >> 9;                           // D_ == 512
  const int k = idx & (D_ - 1);
  posT[idx] = (__bf16)pos_W[k * L_ + n];            // coalesced writes
}

// ---------------------------------------------------------------------------
// Kernel 2: logits = feats @ pos_W + pos_b ; cls = argmax(logits)  [WMMA bf16]
// One wave per 16-row tile: 4 accumulators (N = 64 = 4 x 16). Per the ISA
// fragment layouts, lane (r,h)'s 16 A/B elements are two contiguous 8-element
// K-runs -> all operands fetched with b128 loads.
// ---------------------------------------------------------------------------
__global__ __launch_bounds__(256) void cls_kernel(
    const float* __restrict__ fb, const __bf16* __restrict__ posT,
    const float* __restrict__ pos_b, int* __restrict__ cls) {
  const int lane = threadIdx.x & 31;
  const int wave = threadIdx.x >> 5;
  const int tile = blockIdx.x * 8 + wave;   // 512 tiles total
  const int rowbase = tile * 16;
  const int r  = lane & 15;        // A: row in tile ; B/C: column in N-tile
  const int h8 = (lane >> 4) * 8;  // half-wave K sub-offset per ISA layout

  v8f acc0 = {}, acc1 = {}, acc2 = {}, acc3 = {};
  const float*  __restrict__ arow = fb + (size_t)(rowbase + r) * D_ + h8;
  const __bf16* __restrict__ bp0 = posT + (size_t)(0 * 16 + r) * D_ + h8;
  const __bf16* __restrict__ bp1 = posT + (size_t)(1 * 16 + r) * D_ + h8;
  const __bf16* __restrict__ bp2 = posT + (size_t)(2 * 16 + r) * D_ + h8;
  const __bf16* __restrict__ bp3 = posT + (size_t)(3 * 16 + r) * D_ + h8;

  for (int kk = 0; kk < D_; kk += 32) {
    // A fragment: floats [kk, +8) and [kk+16, +8) relative to arow (pre-offset
    // by h8) -> four b128 loads, packed f32->bf16 converts.
    const float4 x0 = *(const float4*)(arow + kk);
    const float4 x1 = *(const float4*)(arow + kk + 4);
    const float4 x2 = *(const float4*)(arow + kk + 16);
    const float4 x3 = *(const float4*)(arow + kk + 20);
    v16bf a;
    a[0]  = (__bf16)x0.x; a[1]  = (__bf16)x0.y; a[2]  = (__bf16)x0.z; a[3]  = (__bf16)x0.w;
    a[4]  = (__bf16)x1.x; a[5]  = (__bf16)x1.y; a[6]  = (__bf16)x1.z; a[7]  = (__bf16)x1.w;
    a[8]  = (__bf16)x2.x; a[9]  = (__bf16)x2.y; a[10] = (__bf16)x2.z; a[11] = (__bf16)x2.w;
    a[12] = (__bf16)x3.x; a[13] = (__bf16)x3.y; a[14] = (__bf16)x3.z; a[15] = (__bf16)x3.w;

    // B fragments from the bf16 column-major copy: two b128 loads each.
    const v16bf bm0 = load_frag_bf16(bp0 + kk);
    const v16bf bm1 = load_frag_bf16(bp1 + kk);
    const v16bf bm2 = load_frag_bf16(bp2 + kk);
    const v16bf bm3 = load_frag_bf16(bp3 + kk);

    acc0 = __builtin_amdgcn_wmma_f32_16x16x32_bf16(false, a, false, bm0, (short)0, acc0, false, false);
    acc1 = __builtin_amdgcn_wmma_f32_16x16x32_bf16(false, a, false, bm1, (short)0, acc1, false, false);
    acc2 = __builtin_amdgcn_wmma_f32_16x16x32_bf16(false, a, false, bm2, (short)0, acc2, false, false);
    acc3 = __builtin_amdgcn_wmma_f32_16x16x32_bf16(false, a, false, bm3, (short)0, acc3, false, false);
  }

  // Epilogue: +pos_b, per-row argmax. C layout: VGPR m -> row (m + h*8),
  // col = (lane&15) + 16*t. Butterfly over the 16 lanes sharing h.
  const float pb0 = pos_b[r], pb1 = pos_b[16 + r], pb2 = pos_b[32 + r], pb3 = pos_b[48 + r];
#pragma unroll
  for (int m = 0; m < 8; ++m) {
    const float v0 = acc0[m] + pb0, v1 = acc1[m] + pb1;
    const float v2 = acc2[m] + pb2, v3 = acc3[m] + pb3;
    float bv = v0; int bc = r;                       // ascending col, strict >
    if (v1 > bv) { bv = v1; bc = 16 + r; }           // keeps first max (jnp)
    if (v2 > bv) { bv = v2; bc = 32 + r; }
    if (v3 > bv) { bv = v3; bc = 48 + r; }
#pragma unroll
    for (int msk = 1; msk < 16; msk <<= 1) {
      const float ov = __shfl_xor(bv, msk, 32);
      const int   oc = __shfl_xor(bc, msk, 32);
      if (ov > bv || (ov == bv && oc < bc)) { bv = ov; bc = oc; }
    }
    if (r == 0) cls[rowbase + (h8 >> 3) * 8 + m] = bc;
  }
}

// ---------------------------------------------------------------------------
// Kernel 3: sequential greedy parse. One 512-thread workgroup per batch.
// Adjacency = 512x512 bitset in LDS (32 KB). Per step:
//   per-row best j via ctz bit-walk, block argmax (smaller flat index wins),
//   parent = f[i]@W1a + f[j]@W1b + b1 (coalesced W1 row reads, L2-resident,
//   per-lane global_prefetch to pull upcoming rows into near caches),
//   si[i]/sj[i] rank-1 refresh, adjacency row/col merge.
// ---------------------------------------------------------------------------
__global__ __launch_bounds__(512) void parse_kernel(
    const float* __restrict__ W1, const float* __restrict__ b1,
    const float* __restrict__ uvec, const float* __restrict__ vvec,
    const float* __restrict__ cscal, const int* __restrict__ cls,
    float* __restrict__ fb, float* __restrict__ out) {
  __shared__ unsigned adj[S_ * ADJW];           // 32 KB
  __shared__ float sjv[S_], siv[S_];
  __shared__ float rv[S_];
  __shared__ int   ri[S_], rj[S_];
  __shared__ float fi_s[S_], fj_s[S_];
  __shared__ float u_s[S_], v_s[S_];
  __shared__ int   clsl[S_];
  __shared__ unsigned tmp[ADJW];
  __shared__ int   sel_i, sel_j;
  __shared__ float sel_v;

  const int t = threadIdx.x;                     // 0..511
  const int b = blockIdx.x;
  float* __restrict__ f = fb + (size_t)b * S_ * D_;
  const float cconst = cscal[0];

  u_s[t] = uvec[t];
  v_s[t] = vvec[t];
  clsl[t] = cls[b * S_ + t];
  __syncthreads();

  // Build adjacency: (|i-j|==1 | cls equal) & i!=j
  {
    const int cr = clsl[t];
    for (int w = 0; w < ADJW; ++w) {
      unsigned bits = 0u;
      const int jb = w * 32;
#pragma unroll 8
      for (int bi = 0; bi < 32; ++bi) {
        const int j = jb + bi;
        const bool e = (j != t) && ((clsl[j] == cr) | (j == t - 1) | (j == t + 1));
        bits |= e ? (1u << bi) : 0u;
      }
      adj[t * ADJW + w] = bits;
    }
  }

  // Initial si/sj: one wave per 32 rows, coalesced f reads + shuffle reduce
  {
    const int lane = t & 31, wave = t >> 5;
    for (int rr = wave * 32; rr < wave * 32 + 32; ++rr) {
      const float* __restrict__ frow = f + (size_t)rr * D_;
      float du = 0.f, dv = 0.f;
      for (int k = lane; k < D_; k += 32) {
        const float x = frow[k];
        du = fmaf(x, u_s[k], du);
        dv = fmaf(x, v_s[k], dv);
      }
#pragma unroll
      for (int o = 16; o > 0; o >>= 1) {
        du += __shfl_down(du, o, 32);
        dv += __shfl_down(dv, o, 32);
      }
      if (lane == 0) { siv[rr] = du; sjv[rr] = dv; }
    }
  }
  __syncthreads();

  float tot = 0.f;
  for (int step = 0; step < NSTEP; ++step) {
    // ---- per-row best j over adjacency bits (sparse ctz walk) ----
    {
      float best = -INFINITY;
      int bj = 0;
#pragma unroll
      for (int w = 0; w < ADJW; ++w) {
        unsigned bits = adj[t * ADJW + w];
        while (bits) {
          const int bi = __builtin_ctz(bits);
          bits &= bits - 1u;
          const int j = w * 32 + bi;
          const float sv = sjv[j];
          if (sv > best) { best = sv; bj = j; }   // strict > keeps smallest j
        }
      }
      rv[t] = siv[t] + best;                       // -inf for dead rows
      ri[t] = t;
      rj[t] = bj;
    }
    __syncthreads();
    // ---- block argmax; tie -> smaller i (row-major flat argmax) ----
    for (int s = 256; s > 0; s >>= 1) {
      if (t < s) {
        const float vo = rv[t + s];
        if (vo > rv[t] || (vo == rv[t] && ri[t + s] < ri[t])) {
          rv[t] = vo; ri[t] = ri[t + s]; rj[t] = rj[t + s];
        }
      }
      __syncthreads();
    }
    if (t == 0) { sel_i = ri[0]; sel_j = rj[0]; sel_v = rv[0] + cconst; }
    __syncthreads();
    const int i = sel_i, j = sel_j;
    if (t == 0) {
      tot += sel_v;
      out[B_ + ((size_t)b * NSTEP + step) * 2 + 0] = (float)i;
      out[B_ + ((size_t)b * NSTEP + step) * 2 + 1] = (float)j;
    }

    // ---- stage f[i], f[j] into LDS ----
    fi_s[t] = f[(size_t)i * D_ + t];
    fj_s[t] = f[(size_t)j * D_ + t];
    __syncthreads();

    // ---- parent[d] = b1[d] + sum_k fi[k]*W1a[k,d] + fj[k]*W1b[k,d] ----
    float acc = b1[t];
    {
      const float* __restrict__ Wa = W1 + t;                    // W1[k*D + d]
      const float* __restrict__ Wb = W1 + (size_t)D_ * D_ + t;  // W1[(D+k)*D + d]
      for (int k0 = 0; k0 < D_; k0 += 32) {
        if (k0 + 32 < D_) {
          // Per-lane prefetch of upcoming W1 rows: 512 lanes cover the whole
          // 2 KB row pair -> emits global_prefetch into near caches.
          __builtin_prefetch(Wa + (size_t)(k0 + 32) * D_, 0, 1);
          __builtin_prefetch(Wb + (size_t)(k0 + 32) * D_, 0, 1);
        }
#pragma unroll 8
        for (int k = k0; k < k0 + 32; ++k) {
          acc = fmaf(fi_s[k], Wa[(size_t)k * D_], acc);
          acc = fmaf(fj_s[k], Wb[(size_t)k * D_], acc);
        }
      }
    }
    __syncthreads();                       // fi_s/fj_s free for reuse

    f[(size_t)i * D_ + t] = acc;           // f[i] <- parent
    rv[t]   = acc * u_s[t];                // refresh si[i], sj[i]
    fi_s[t] = acc * v_s[t];
    __syncthreads();
    for (int s = 256; s > 0; s >>= 1) {
      if (t < s) { rv[t] += rv[t + s]; fi_s[t] += fi_s[t + s]; }
      __syncthreads();
    }
    if (t == 0) { siv[i] = rv[0]; sjv[i] = fi_s[0]; }

    // ---- adjacency merge: row = a[i]|a[j] minus bits {i,j} ----
    if (t < ADJW) {
      unsigned m = adj[i * ADJW + t] | adj[j * ADJW + t];
      if (t == (i >> 5)) m &= ~(1u << (i & 31));
      if (t == (j >> 5)) m &= ~(1u << (j & 31));
      tmp[t] = m;
    }
    __syncthreads();
    if (t == i) {
      for (int w = 0; w < ADJW; ++w) adj[i * ADJW + w] = tmp[w];
    } else if (t == j) {
      for (int w = 0; w < ADJW; ++w) adj[j * ADJW + w] = 0u;
    } else {
      const unsigned has = (tmp[t >> 5] >> (t & 31)) & 1u;
      unsigned wi = adj[t * ADJW + (i >> 5)];
      wi = (wi & ~(1u << (i & 31))) | (has << (i & 31));
      adj[t * ADJW + (i >> 5)] = wi;
      adj[t * ADJW + (j >> 5)] &= ~(1u << (j & 31));
    }
    __syncthreads();
  }

  if (t == 0) out[b] = tot;
}

// ---------------------------------------------------------------------------
// Launch: uvc | gather | posT -> WMMA cls -> per-batch parse. Workspace:
//   [0, 16MB)          feats (B*S*D f32)  -- fully rewritten each launch
//   [16MB, +32KB)      cls   (B*S i32)
//   [16MB+32KB, +64KB) posT  (L x D bf16, column-major)
//   then u[512], v[512], c[1]
// ---------------------------------------------------------------------------
extern "C" void kernel_launch(void* const* d_in, const int* in_sizes, int n_in,
                              void* d_out, int out_size, void* d_ws, size_t ws_size,
                              hipStream_t stream) {
  (void)in_sizes; (void)n_in; (void)out_size; (void)ws_size;
  const int*   tok   = (const int*)  d_in[0];
  const float* emb   = (const float*)d_in[1];
  const float* pos_W = (const float*)d_in[2];
  const float* pos_b = (const float*)d_in[3];
  const float* W1    = (const float*)d_in[4];
  const float* b1    = (const float*)d_in[5];
  const float* w2    = (const float*)d_in[6];
  const float* b2    = (const float*)d_in[7];
  float* out = (float*)d_out;

  char* ws = (char*)d_ws;
  const size_t fb_bytes   = (size_t)B_ * S_ * D_ * sizeof(float);   // 16 MB
  const size_t cls_bytes  = (size_t)B_ * S_ * sizeof(int);          // 32 KB
  const size_t posT_bytes = (size_t)L_ * D_ * sizeof(__bf16);       // 64 KB

  float*  fb   = (float*)ws;
  int*    cls  = (int*)  (ws + fb_bytes);
  __bf16* posT = (__bf16*)(ws + fb_bytes + cls_bytes);
  float*  u    = (float*)(ws + fb_bytes + cls_bytes + posT_bytes);
  float*  v    = u + D_;
  float*  cbuf = v + D_;

  uvc_kernel   <<<1, 512, 0, stream>>>(W1, b1, w2, b2, u, v, cbuf);
  gather_kernel<<<(B_ * S_ * D_) / 256, 256, 0, stream>>>(tok, emb, fb);
  posT_kernel  <<<(D_ * L_) / 256, 256, 0, stream>>>(pos_W, posT);
  cls_kernel   <<<(B_ * S_ / 16) / 8, 256, 0, stream>>>(fb, posT, pos_b, cls);
  parse_kernel <<<B_, 512, 0, stream>>>(W1, b1, u, v, cbuf, cls, fb, out);
}
